// FLASHDecoderLayer_49065706390114
// MI455X (gfx1250) — compile-verified, matrix operand order checked by reading, not run on hardware
//
#include <hip/hip_runtime.h>

#define N_TOKENS 4096
#define HIDDEN   2048
#define INTER    1024
#define NEXP     8
#define NROUTE   10   // 8 real + 2 zero experts
#define SCALING  2.5f
#define BK       32
#define LDA      40   // padded LDS row stride (bf16 elements)

typedef __bf16 bf16;
typedef __attribute__((ext_vector_type(16))) __bf16 v16bf;
typedef __attribute__((ext_vector_type(8)))  float  v8f;

union Frag16 { uint4 u[2]; bf16 h[16]; v16bf v; };
union Frag8  { uint4 u;    bf16 h[8]; };

__device__ __forceinline__ v16bf ldsFrag(const bf16* p) {
  Frag16 f;
  f.u[0] = *(const uint4*)(p);
  f.u[1] = *(const uint4*)(p + 8);
  return f.v;
}

__device__ __forceinline__ v8f wmma_bf16(v16bf a, v16bf b, v8f c) {
  return __builtin_amdgcn_wmma_f32_16x16x32_bf16(
      /*neg_a=*/false, a, /*neg_b=*/false, b,
      /*c_mod=*/(short)0, c, /*reuse_a=*/false, /*reuse_b=*/false);
}

// convert 16 consecutive f32 -> 16 bf16, store 32B to LDS (16B-aligned dst)
__device__ __forceinline__ void cvt16_store(const float4* f, bf16* dst) {
  Frag16 fr;
#pragma unroll
  for (int i = 0; i < 4; ++i) {
    fr.h[4*i+0] = (bf16)f[i].x; fr.h[4*i+1] = (bf16)f[i].y;
    fr.h[4*i+2] = (bf16)f[i].z; fr.h[4*i+3] = (bf16)f[i].w;
  }
  *(uint4*)(dst)     = fr.u[0];
  *(uint4*)(dst + 8) = fr.u[1];
}

__device__ __forceinline__ void cvt8_store(float4 a, float4 b, bf16* dst) {
  Frag8 fr;
  fr.h[0] = (bf16)a.x; fr.h[1] = (bf16)a.y; fr.h[2] = (bf16)a.z; fr.h[3] = (bf16)a.w;
  fr.h[4] = (bf16)b.x; fr.h[5] = (bf16)b.y; fr.h[6] = (bf16)b.z; fr.h[7] = (bf16)b.w;
  *(uint4*)dst = fr.u;
}

__device__ __forceinline__ void load16(const float* p, float4* r) {
  const float4* q = (const float4*)p;
  r[0] = q[0]; r[1] = q[1]; r[2] = q[2]; r[3] = q[3];
}

// CDNA5 async global->LDS copy: 32B per lane (two b128s; offset:16 applies to
// both the LDS and the global address per the TDM/async addressing rules).
__device__ __forceinline__ void async_copy32(unsigned lds_off, const bf16* gptr) {
  asm volatile("global_load_async_to_lds_b128 %0, %1, off\n\t"
               "global_load_async_to_lds_b128 %0, %1, off offset:16"
               :: "v"(lds_off), "v"(gptr) : "memory");
}
__device__ __forceinline__ void wait_async0() {
  asm volatile("s_wait_asynccnt 0x0" ::: "memory");
}

// ---------------------------------------------------------------------------
// Kernel 1: router. One wave32 per token: logits = x . router_w^T (10 experts),
// sigmoid, top-2 (first-index tie-break like jax.lax.top_k), split real vs zero.
// ---------------------------------------------------------------------------
__global__ __launch_bounds__(256) void router_kernel(
    const float* __restrict__ x, const float* __restrict__ rw,
    const float* __restrict__ bias,
    int* __restrict__ tok_e, float* __restrict__ tok_w, float* __restrict__ wz)
{
  const int wave = threadIdx.x >> 5;
  const int lane = threadIdx.x & 31;
  const int t = blockIdx.x * 8 + wave;
  const float* xr = x + (size_t)t * HIDDEN;

  float acc[NROUTE];
#pragma unroll
  for (int e = 0; e < NROUTE; ++e) acc[e] = 0.f;

  for (int k = lane; k < HIDDEN; k += 32) {
    float xv = xr[k];
#pragma unroll
    for (int e = 0; e < NROUTE; ++e) acc[e] += xv * rw[e * HIDDEN + k];
  }
#pragma unroll
  for (int e = 0; e < NROUTE; ++e)
#pragma unroll
    for (int off = 16; off > 0; off >>= 1)
      acc[e] += __shfl_xor(acc[e], off, 32);

  if (lane == 0) {
    float s[NROUTE], sel[NROUTE];
#pragma unroll
    for (int e = 0; e < NROUTE; ++e) {
      s[e] = 1.f / (1.f + __expf(-acc[e]));
      sel[e] = s[e] + bias[e];
    }
    int i1 = 0; float b1 = sel[0];
#pragma unroll
    for (int e = 1; e < NROUTE; ++e) if (sel[e] > b1) { b1 = sel[e]; i1 = e; }
    int i2 = -1; float b2 = -3.4e38f;
#pragma unroll
    for (int e = 0; e < NROUTE; ++e)
      if (e != i1 && sel[e] > b2) { b2 = sel[e]; i2 = e; }

    float wzero = 0.f;
    int ids[2] = { i1, i2 };
#pragma unroll
    for (int j = 0; j < 2; ++j) {
      int id = ids[j];
      if (id < NEXP) { tok_e[2*t + j] = id; tok_w[2*t + j] = s[id]; }
      else           { tok_e[2*t + j] = -1; tok_w[2*t + j] = 0.f; wzero += s[id]; }
    }
    wz[t] = wzero;
  }
}

// ---------------------------------------------------------------------------
// Kernel 2: deterministic stream compaction. Wave e = expert e. Ballot/popc
// scan over tokens in order -> counts, prefix offsets, gathered token lists.
// ---------------------------------------------------------------------------
__global__ __launch_bounds__(256) void gather_kernel(
    const int* __restrict__ tok_e, const float* __restrict__ tok_w,
    int* __restrict__ counts, int* __restrict__ offsets,
    int* __restrict__ g_tok, float* __restrict__ g_w)
{
  __shared__ int s_cnt[NEXP], s_off[NEXP];
  const int wave = threadIdx.x >> 5;   // expert id, 0..7
  const int lane = threadIdx.x & 31;

  int cnt = 0;
  for (int tb = 0; tb < N_TOKENS; tb += 32) {
    int t = tb + lane;
    bool sel = (tok_e[2*t] == wave) || (tok_e[2*t + 1] == wave);
    unsigned long long mask = __ballot(sel);
    cnt += __popcll(mask);
  }
  if (lane == 0) s_cnt[wave] = cnt;
  __syncthreads();
  if (threadIdx.x == 0) {
    int run = 0;
    for (int e = 0; e < NEXP; ++e) {
      s_off[e] = run; offsets[e] = run; counts[e] = s_cnt[e]; run += s_cnt[e];
    }
  }
  __syncthreads();

  int pos = s_off[wave];
  for (int tb = 0; tb < N_TOKENS; tb += 32) {
    int t = tb + lane;
    int j = -1;
    if      (tok_e[2*t]     == wave) j = 0;
    else if (tok_e[2*t + 1] == wave) j = 1;
    unsigned long long mask = __ballot(j >= 0);
    if (j >= 0) {
      int my = pos + __popcll(mask & ((1ull << lane) - 1ull));
      g_tok[my] = t;
      g_w[my]   = tok_w[2*t + j];
    }
    pos += __popcll(mask);
  }
}

// ---------------------------------------------------------------------------
// Kernel 3: out = SCALING * w_zero[t] * x  (also initializes the output)
// ---------------------------------------------------------------------------
__global__ __launch_bounds__(256) void init_out_kernel(
    const float* __restrict__ x, const float* __restrict__ wz,
    float* __restrict__ out)
{
  size_t i = (size_t)blockIdx.x * 256 + threadIdx.x;
  int t = (int)(i >> 11);  // HIDDEN == 2048
  out[i] = SCALING * wz[t] * x[i];
}

// ---------------------------------------------------------------------------
// Kernel 4: gate_up GEMM fused with SiLU.  Double-buffered LDS pipeline:
// global loads for tile k+1 are issued before tile k's WMMAs, one barrier
// per K-step. Out-of-range token rows are zeroed in LDS once, before the
// loop, so the steady state has no zero-fill VALU work.
// 8 waves (4 M x 2 N), 8 v_wmma_f32_16x16x32_bf16 per K-step.
// ---------------------------------------------------------------------------
__global__ __launch_bounds__(256) void gemm1_kernel(
    const float* __restrict__ x, const float* __restrict__ wgu,
    const int* __restrict__ counts, const int* __restrict__ offsets,
    const int* __restrict__ g_tok, bf16* __restrict__ hbuf)
{
  const int e   = blockIdx.z;
  const int cnt = counts[e];
  const int m0  = blockIdx.y * 128;
  if (m0 >= cnt) return;
  const int n0   = blockIdx.x * 64;
  const int base = offsets[e];

  __shared__ bf16 As[2][128 * LDA];
  __shared__ bf16 Bs[2][128 * LDA];   // rows 0..63: gate, rows 64..127: up

  const int tid  = threadIdx.x;
  const int lane = tid & 31;
  const int wave = tid >> 5;
  const int wm   = wave & 3;
  const int wn   = wave >> 2;
  const int hs   = (lane >> 4) << 4;     // K-half select: 0 or 16
  const int fr   = lane & 15;
  const int mlo  = (lane >> 4) << 3;

  const int lrow = tid >> 1;
  const int lcol = (tid & 1) << 4;
  const bool avalid = (m0 + lrow) < cnt;
  const float* aptr = avalid
      ? (x + (size_t)g_tok[base + m0 + lrow] * HIDDEN + lcol) : nullptr;
  const int gf = (lrow < 64) ? (n0 + lrow) : (INTER + n0 + (lrow - 64));
  const float* bptr = wgu + ((size_t)e * (2 * INTER) + gf) * HIDDEN + lcol;

  // zero-fill out-of-range token rows once; they are never overwritten
  if (!avalid) {
    uint4 z = make_uint4(0, 0, 0, 0);
    *(uint4*)&As[0][lrow * LDA + lcol]     = z;
    *(uint4*)&As[0][lrow * LDA + lcol + 8] = z;
    *(uint4*)&As[1][lrow * LDA + lcol]     = z;
    *(uint4*)&As[1][lrow * LDA + lcol + 8] = z;
  }

  v8f cg[2][2] = {{{0.f},{0.f}},{{0.f},{0.f}}};
  v8f cu[2][2] = {{{0.f},{0.f}},{{0.f},{0.f}}};

  float4 ra[4], rb[4];
  if (avalid) {
    load16(aptr, ra);
    cvt16_store(ra, &As[0][lrow * LDA + lcol]);
  }
  load16(bptr, rb);
  cvt16_store(rb, &Bs[0][lrow * LDA + lcol]);
  __syncthreads();

  const int NK = HIDDEN / BK;
  int cur = 0;
  for (int kk = 0; kk < NK; ++kk) {
    const bool hasNext = (kk + 1) < NK;
    if (hasNext) {
      if (avalid) load16(aptr + (kk + 1) * BK, ra);
      load16(bptr + (kk + 1) * BK, rb);
    }

    v16bf a0  = ldsFrag(&As[cur][(wm * 32 +      fr) * LDA + hs]);
    v16bf a1  = ldsFrag(&As[cur][(wm * 32 + 16 + fr) * LDA + hs]);
    v16bf bg0 = ldsFrag(&Bs[cur][(wn * 32 +      fr) * LDA + hs]);
    v16bf bg1 = ldsFrag(&Bs[cur][(wn * 32 + 16 + fr) * LDA + hs]);
    v16bf bu0 = ldsFrag(&Bs[cur][(64 + wn * 32 +      fr) * LDA + hs]);
    v16bf bu1 = ldsFrag(&Bs[cur][(64 + wn * 32 + 16 + fr) * LDA + hs]);

    cg[0][0] = wmma_bf16(a0, bg0, cg[0][0]);
    cg[0][1] = wmma_bf16(a0, bg1, cg[0][1]);
    cg[1][0] = wmma_bf16(a1, bg0, cg[1][0]);
    cg[1][1] = wmma_bf16(a1, bg1, cg[1][1]);
    cu[0][0] = wmma_bf16(a0, bu0, cu[0][0]);
    cu[0][1] = wmma_bf16(a0, bu1, cu[0][1]);
    cu[1][0] = wmma_bf16(a1, bu0, cu[1][0]);
    cu[1][1] = wmma_bf16(a1, bu1, cu[1][1]);

    if (hasNext) {
      const int nxt = cur ^ 1;
      if (avalid) cvt16_store(ra, &As[nxt][lrow * LDA + lcol]);
      cvt16_store(rb, &Bs[nxt][lrow * LDA + lcol]);
      __syncthreads();
      cur = nxt;
    }
  }

#pragma unroll
  for (int i = 0; i < 2; ++i)
#pragma unroll
    for (int j = 0; j < 2; ++j) {
#pragma unroll
      for (int r = 0; r < 8; ++r) {
        int m = m0 + wm * 32 + i * 16 + r + mlo;
        if (m < cnt) {
          float g = cg[i][j][r];
          float u = cu[i][j][r];
          float h = g / (1.f + __expf(-g)) * u;   // silu(g) * u
          int n = n0 + wn * 32 + j * 16 + fr;
          hbuf[(size_t)(base + m) * INTER + n] = (bf16)h;
        }
      }
    }
}

// ---------------------------------------------------------------------------
// Kernel 5: down GEMM. A operand (hbuf) is already bf16 -> moved with CDNA5
// GLOBAL_LOAD_ASYNC_TO_LDS_B128 (ASYNCcnt), no VGPR round-trip. B (fp32
// weights) reg-staged + cvt. Double-buffered LDS, one barrier per K-step.
// Epilogue scatters 2.5 * topk_w * y into out via global_atomic_add_f32.
// ---------------------------------------------------------------------------
__global__ __launch_bounds__(256) void gemm2_kernel(
    const bf16* __restrict__ hbuf, const float* __restrict__ wdn,
    const int* __restrict__ counts, const int* __restrict__ offsets,
    const int* __restrict__ g_tok, const float* __restrict__ g_w,
    float* __restrict__ out)
{
  const int e   = blockIdx.z;
  const int cnt = counts[e];
  const int m0  = blockIdx.y * 128;
  if (m0 >= cnt) return;
  const int n0   = blockIdx.x * 64;
  const int base = offsets[e];

  __shared__ bf16 As[2][128 * LDA];
  __shared__ bf16 Bs[2][64 * LDA];

  const int tid  = threadIdx.x;
  const int lane = tid & 31;
  const int wave = tid >> 5;
  const int wm   = wave & 3;
  const int wn   = wave >> 2;
  const int hs   = (lane >> 4) << 4;
  const int fr   = lane & 15;
  const int mlo  = (lane >> 4) << 3;

  const int arow = tid >> 1;
  const int acol = (tid & 1) << 4;
  const bool avalid = (m0 + arow) < cnt;
  const bf16* aptr = hbuf + (size_t)(base + m0 + arow) * INTER + acol;
  const int brow = tid >> 2;
  const int bcol = (tid & 3) << 3;
  const float* bptr = wdn + ((size_t)e * HIDDEN + n0 + brow) * INTER + bcol;

  // per-lane LDS byte offsets for the async copies (low 32 bits of the
  // generic pointer = wave-relative LDS address)
  unsigned ldsA[2];
  ldsA[0] = (unsigned)(size_t)(void*)&As[0][arow * LDA + acol];
  ldsA[1] = (unsigned)(size_t)(void*)&As[1][arow * LDA + acol];

  // zero-fill out-of-range rows once (they are never overwritten)
  if (!avalid) {
    uint4 z = make_uint4(0, 0, 0, 0);
    *(uint4*)&As[0][arow * LDA + acol]     = z;
    *(uint4*)&As[0][arow * LDA + acol + 8] = z;
    *(uint4*)&As[1][arow * LDA + acol]     = z;
    *(uint4*)&As[1][arow * LDA + acol + 8] = z;
  }

  v8f acc[2][2] = {{{0.f},{0.f}},{{0.f},{0.f}}};

  // prologue: tile 0
  if (avalid) async_copy32(ldsA[0], aptr);
  float4 rb0 = *(const float4*)(bptr);
  float4 rb1 = *(const float4*)(bptr + 4);
  cvt8_store(rb0, rb1, &Bs[0][brow * LDA + bcol]);
  wait_async0();
  __syncthreads();

  const int NK = INTER / BK;
  int cur = 0;
  for (int kk = 0; kk < NK; ++kk) {
    const bool hasNext = (kk + 1) < NK;
    if (hasNext) {
      if (avalid) async_copy32(ldsA[cur ^ 1], aptr + (kk + 1) * BK);
      rb0 = *(const float4*)(bptr + (kk + 1) * BK);
      rb1 = *(const float4*)(bptr + (kk + 1) * BK + 4);
    }

    v16bf a0  = ldsFrag(&As[cur][(wm * 32 +      fr) * LDA + hs]);
    v16bf a1  = ldsFrag(&As[cur][(wm * 32 + 16 + fr) * LDA + hs]);
    v16bf b0f = ldsFrag(&Bs[cur][(wn * 32 +      fr) * LDA + hs]);
    v16bf b1f = ldsFrag(&Bs[cur][(wn * 32 + 16 + fr) * LDA + hs]);

    acc[0][0] = wmma_bf16(a0, b0f, acc[0][0]);
    acc[0][1] = wmma_bf16(a0, b1f, acc[0][1]);
    acc[1][0] = wmma_bf16(a1, b0f, acc[1][0]);
    acc[1][1] = wmma_bf16(a1, b1f, acc[1][1]);

    if (hasNext) {
      const int nxt = cur ^ 1;
      cvt8_store(rb0, rb1, &Bs[nxt][brow * LDA + bcol]);
      wait_async0();          // our async writes into As[nxt] have landed
      __syncthreads();        // everyone's LDS writes visible
      cur = nxt;
    }
  }

#pragma unroll
  for (int i = 0; i < 2; ++i)
#pragma unroll
    for (int j = 0; j < 2; ++j) {
#pragma unroll
      for (int r = 0; r < 8; ++r) {
        int m = m0 + wm * 32 + i * 16 + r + mlo;
        if (m < cnt) {
          int slot = base + m;
          float wgt = g_w[slot] * SCALING;
          int tok   = g_tok[slot];
          int n = n0 + wn * 32 + j * 16 + fr;
          atomicAdd(out + (size_t)tok * HIDDEN + n, acc[i][j][r] * wgt);
        }
      }
    }
}

// ---------------------------------------------------------------------------
extern "C" void kernel_launch(void* const* d_in, const int* in_sizes, int n_in,
                              void* d_out, int out_size, void* d_ws, size_t ws_size,
                              hipStream_t stream) {
  const float* x    = (const float*)d_in[0];  // (4096, 2048)
  const float* rw   = (const float*)d_in[1];  // (10, 2048)
  const float* bias = (const float*)d_in[2];  // (10,)
  const float* wgu  = (const float*)d_in[3];  // (8, 2048, 2048)
  const float* wdn  = (const float*)d_in[4];  // (8, 2048, 1024)
  float* out = (float*)d_out;                 // (4096, 2048)

  char* w = (char*)d_ws;
  int*   counts  = (int*)w;   w += 256;
  int*   offsets = (int*)w;   w += 256;
  int*   tok_e   = (int*)w;   w += (size_t)N_TOKENS * 2 * sizeof(int);
  float* tok_w   = (float*)w; w += (size_t)N_TOKENS * 2 * sizeof(float);
  float* wz      = (float*)w; w += (size_t)N_TOKENS * sizeof(float);
  int*   g_tok   = (int*)w;   w += (size_t)N_TOKENS * 2 * sizeof(int);
  float* g_w     = (float*)w; w += (size_t)N_TOKENS * 2 * sizeof(float);
  w = (char*)(((size_t)w + 255) & ~(size_t)255);
  bf16*  hbuf    = (bf16*)w;  // 8192 * 1024 bf16 = 16 MiB

  router_kernel<<<N_TOKENS / 8, 256, 0, stream>>>(x, rw, bias, tok_e, tok_w, wz);
  gather_kernel<<<1, 256, 0, stream>>>(tok_e, tok_w, counts, offsets, g_tok, g_w);
  init_out_kernel<<<(N_TOKENS * HIDDEN) / 256, 256, 0, stream>>>(x, wz, out);
  gemm1_kernel<<<dim3(INTER / 64, N_TOKENS / 128, NEXP), 256, 0, stream>>>(
      x, wgu, counts, offsets, g_tok, hbuf);
  gemm2_kernel<<<dim3(HIDDEN / 64, N_TOKENS / 128, NEXP), 256, 0, stream>>>(
      hbuf, wdn, counts, offsets, g_tok, g_w, out);
}
